// MultiTaskLSTM_87531433493188
// MI455X (gfx1250) — matrix operand order repaired and checked by reading, not compiled, not run on hardware
//
#include <hip/hip_runtime.h>
#include <hip/hip_bf16.h>

// ---------------------------------------------------------------------------
// MultiTaskLSTM for MI455X (gfx1250, wave32, WMMA).
// Sizes from reference: B=1024, T_IN=168, T_OUT=24, F=8, H=256.
// ---------------------------------------------------------------------------

#define Bq    1024
#define TIN   168
#define TOUT  24
#define Fq    8
#define Hq    256
#define G4    1024      // 4*H
#define BT    16        // batch tile (one WMMA M-tile)
#define NBT   (Bq/BT)   // 64 batch tiles

typedef __attribute__((ext_vector_type(16))) __bf16 v16bf;
typedef __attribute__((ext_vector_type(8)))  float  v8f;

union Frag { unsigned int u[8]; v16bf v; };

// B-fragment chunk size used for software pipelining (keeps VGPR pressure low:
// b[2][CH] <= 64 VGPRs instead of b[2][NSUB] = 128 which spilled in round 2).
constexpr int chunk_of(int nsub) {
  return (nsub % 4 == 0) ? 4 : ((nsub % 3 == 0) ? 3 : nsub);
}

// ---- fragment index maps (contiguous k = 16*(lane/16) + 2j + e for both A and B, wave32) -------------------
__device__ __forceinline__ int frag_kA(int lane, int j, int e) {
  return ((lane >> 4) << 4) + (j << 1) + e;
}
__device__ __forceinline__ int frag_kB(int lane, int j, int e) {
  return ((lane >> 4) << 4) + (j << 1) + e;
}

// ---- bf16 helpers ---------------------------------------------------------
__device__ __forceinline__ unsigned short f2bf(float f) {
  unsigned int u = __builtin_bit_cast(unsigned int, f);
  u += 0x7fffu + ((u >> 16) & 1u);        // round-to-nearest-even
  return (unsigned short)(u >> 16);
}
__device__ __forceinline__ unsigned int pack2(float lo, float hi) {
  return (unsigned int)f2bf(lo) | ((unsigned int)f2bf(hi) << 16);
}
__device__ __forceinline__ float bf2f(unsigned short h) {
  unsigned int u = ((unsigned int)h) << 16;
  return __builtin_bit_cast(float, u);
}
__device__ __forceinline__ float sigf(float x)  { return 1.f / (1.f + __expf(-x)); }
__device__ __forceinline__ float leaky(float x) { return x > 0.f ? x : 0.01f * x; }

__device__ __forceinline__ v8f vzero8() {
  v8f z = {0.f,0.f,0.f,0.f,0.f,0.f,0.f,0.f};
  return z;
}

__device__ __forceinline__ v8f wmma_bf16(const Frag& a, const Frag& b, v8f c) {
  return __builtin_amdgcn_wmma_f32_16x16x32_bf16(false, a.v, false, b.v,
                                                 (short)0, c, false, false);
}

// ---------------------------------------------------------------------------
// Weight panel packing (fp32 -> bf16 B-fragment panels).
// Panel layout: uint out[((kt*NT + nt)*32 + lane)*8 + j]  (one uint = 2 bf16)
// ---------------------------------------------------------------------------
__device__ __forceinline__ float fetchW(const float* Wa, int Ka,
                                        const float* Wb, int Kb,
                                        int n, int k) {
  if (k < Ka) return Wa[n * Ka + k];
  k -= Ka;
  if (k < Kb) return Wb[n * Kb + k];
  return 0.f;
}

__global__ void pack_panels_kernel(const float* __restrict__ Wa, int Ka,
                                   const float* __restrict__ Wb, int Kb,
                                   long strideB,
                                   unsigned int* __restrict__ outp, long strideOut,
                                   int NT, int KT) {
  int f = blockIdx.y;
  const float* wb = Wb + (long)f * strideB;
  unsigned int* out = outp + (long)f * strideOut;
  int idx = blockIdx.x * blockDim.x + threadIdx.x;
  int total = KT * NT * 256;
  if (idx >= total) return;
  int j    = idx & 7;
  int lane = (idx >> 3) & 31;
  int tile = idx >> 8;
  int nt = tile % NT, kt = tile / NT;
  int n  = nt * 16 + (lane & 15);
  int k0 = kt * 32 + frag_kB(lane, j, 0);
  float f0 = fetchW(Wa, Ka, wb, Kb, n, k0);
  float f1 = fetchW(Wa, Ka, wb, Kb, n, k0 + 1);
  out[idx] = pack2(f0, f1);
}

// ---------------------------------------------------------------------------
// Software-pipelined wave GEMM core.
//   A: 16 x KT*32 bf16 in LDS (reloaded per kt; DS latency is short).
//   B: loaded by `loadB(Frag bf[CH], kt, s0)` = CH fragments s0..s0+CH-1.
// Chunked 2-deep ping-pong: loads for chunk p+1 are issued before the WMMA
// block of chunk p. All register-array indices are compile-time constants.
// ---------------------------------------------------------------------------
template <int NSUB, typename BL>
__device__ __forceinline__ void gemm_core(const unsigned short* __restrict__ lA,
                                          int ldA, int KT, BL&& loadB, v8f* acc) {
  constexpr int CH = chunk_of(NSUB);
  constexpr int NC = NSUB / CH;
  int lane = threadIdx.x & 31;
  int row  = lane & 15;
  Frag a;
  Frag b[2][CH];
  auto loadA = [&](int kt) {
#pragma unroll
    for (int j = 0; j < 8; ++j)
      a.u[j] = *(const unsigned int*)(lA + row * ldA + kt * 32 + frag_kA(lane, j, 0));
  };
  loadB(b[0], 0, 0);
  if constexpr (NC == 1) {
    // chunk == whole kt; parity from kt (loop unrolled by 2, KT even here)
#pragma unroll 2
    for (int kt = 0; kt < KT; ++kt) {
      loadA(kt);
      int cur = kt & 1, nxt = cur ^ 1;
      if (kt + 1 < KT) loadB(b[nxt], kt + 1, 0);
#pragma unroll
      for (int s = 0; s < CH; ++s)
        acc[s] = wmma_bf16(a, b[cur][s], acc[s]);
    }
  } else {
    // NC chunks per kt (NC even in all uses) -> parity from chunk index only
#pragma unroll 1
    for (int kt = 0; kt < KT; ++kt) {
      loadA(kt);
#pragma unroll
      for (int c = 0; c < NC; ++c) {
        int cur = c & 1, nxt = cur ^ 1;
        bool last = (c == NC - 1);
        if (!(last && (kt + 1 == KT)))
          loadB(b[nxt], last ? kt + 1 : kt, last ? 0 : (c + 1) * CH);
#pragma unroll
        for (int s = 0; s < CH; ++s)
          acc[c * CH + s] = wmma_bf16(a, b[cur][s], acc[c * CH + s]);
      }
    }
  }
}

// Panel-backed B loader (pre-packed weights) + next-slab prefetch.
template <int NSUB>
__device__ __forceinline__ void gemm_panels(const unsigned short* __restrict__ lA,
                                            int ldA, int KT,
                                            const unsigned int* __restrict__ panels,
                                            int NT, int nt0, v8f* acc) {
  constexpr int CH = chunk_of(NSUB);
  int lane = threadIdx.x & 31;
  gemm_core<NSUB>(lA, ldA, KT,
      [&](Frag* bf, int kt, int s0) {
        if (s0 == 0 && kt + 1 < KT)
          __builtin_prefetch(panels + (((long)(kt + 1) * NT + nt0) * 32) * 8, 0, 3);
#pragma unroll
        for (int s = 0; s < CH; ++s) {
          const unsigned int* bp =
              panels + (((long)kt * NT + (nt0 + s0 + s)) * 32 + lane) * 8;
#pragma unroll
          for (int j = 0; j < 8; ++j) bf[s].u[j] = bp[j];
        }
      },
      acc);
}

// Store D tiles (CDNA5 C/D layout: lane -> col = lane&15, rows = 8*(lane>>4)+j)
template <int NSUB>
__device__ __forceinline__ void store_acc_lds(float* __restrict__ g, int ldG,
                                              int nt0, const v8f* acc, float scale) {
  int lane = threadIdx.x & 31;
  int col = lane & 15, mh = (lane >> 4) * 8;
#pragma unroll
  for (int s = 0; s < NSUB; ++s) {
    int n = (nt0 + s) * 16 + col;
#pragma unroll
    for (int j = 0; j < 8; ++j) g[(mh + j) * ldG + n] = acc[s][j] * scale;
  }
}

// ---------------------------------------------------------------------------
// Encoder: persistent WG per 16-row batch tile; both LSTM layers fused per
// timestep; h/c state resident in LDS across all 168 steps.
// ---------------------------------------------------------------------------
__global__ __launch_bounds__(256) void encoder_kernel(
    const float* __restrict__ x,                 // (B, TIN, F)
    const unsigned int* __restrict__ P0,         // layer0 panels: KT=9,  NT=64
    const unsigned int* __restrict__ P1,         // layer1 panels: KT=16, NT=64
    const float* __restrict__ bih0, const float* __restrict__ bhh0,
    const float* __restrict__ bih1, const float* __restrict__ bhh1,
    float* __restrict__ dec_h, float* __restrict__ dec_c,  // (F,B,H)
    float* __restrict__ dec_in)                  // (B,F)
{
  __shared__ unsigned short A0[BT][288];   // [x_t | h0] bf16, padded K=288
  __shared__ unsigned short A1[BT][512];   // [y0  | h1] bf16, K=512
  __shared__ float g[BT][G4];              // gate pre-activations
  __shared__ float c0[BT][Hq];
  __shared__ float c1[BT][Hq];

  int tid = threadIdx.x, wave = tid >> 5;
  int b0 = blockIdx.x * BT;

  for (int i = tid; i < BT * 288; i += 256) (&A0[0][0])[i] = 0;
  for (int i = tid; i < BT * 512; i += 256) (&A1[0][0])[i] = 0;
  for (int i = tid; i < BT * Hq; i += 256) { (&c0[0][0])[i] = 0.f; (&c1[0][0])[i] = 0.f; }
  __syncthreads();

  v8f acc[8];
#pragma unroll 1
  for (int t = 0; t < TIN; ++t) {
    if (tid < BT * Fq) {
      int m = tid >> 3, k = tid & 7;
      A0[m][k] = f2bf(x[((b0 + m) * TIN + t) * Fq + k]);
    }
    __syncthreads();

    // layer 0: g = [x|h0] @ W0^T   (16 x 1024 x 288)
#pragma unroll
    for (int s = 0; s < 8; ++s) acc[s] = vzero8();
    gemm_panels<8>(&A0[0][0], 288, 9, P0, 64, wave * 8, acc);
    store_acc_lds<8>(&g[0][0], G4, wave * 8, acc, 1.f);
    __syncthreads();

    for (int i = tid; i < BT * Hq; i += 256) {
      int m = i >> 8, n = i & 255;
      float gi = g[m][n]       + bih0[n]       + bhh0[n];
      float gf = g[m][256 + n] + bih0[256 + n] + bhh0[256 + n];
      float gg = g[m][512 + n] + bih0[512 + n] + bhh0[512 + n];
      float go = g[m][768 + n] + bih0[768 + n] + bhh0[768 + n];
      float c2 = sigf(gf) * c0[m][n] + sigf(gi) * tanhf(gg);
      float h2 = sigf(go) * tanhf(c2);
      c0[m][n] = c2;
      unsigned short hb = f2bf(h2);
      A1[m][n]     = hb;   // y0 -> layer1 input
      A0[m][8 + n] = hb;   // h0 recurrent
    }
    __syncthreads();

    // layer 1: g = [y0|h1] @ W1^T  (16 x 1024 x 512)
#pragma unroll
    for (int s = 0; s < 8; ++s) acc[s] = vzero8();
    gemm_panels<8>(&A1[0][0], 512, 16, P1, 64, wave * 8, acc);
    store_acc_lds<8>(&g[0][0], G4, wave * 8, acc, 1.f);
    __syncthreads();

    for (int i = tid; i < BT * Hq; i += 256) {
      int m = i >> 8, n = i & 255;
      float gi = g[m][n]       + bih1[n]       + bhh1[n];
      float gf = g[m][256 + n] + bih1[256 + n] + bhh1[256 + n];
      float gg = g[m][512 + n] + bih1[512 + n] + bhh1[512 + n];
      float go = g[m][768 + n] + bih1[768 + n] + bhh1[768 + n];
      float c2 = sigf(gf) * c1[m][n] + sigf(gi) * tanhf(gg);
      float h2 = sigf(go) * tanhf(c2);
      c1[m][n] = c2;
      A1[m][256 + n] = f2bf(h2);   // h1 recurrent
    }
    __syncthreads();
  }

  // decoder init state: h/c mean replicated across F tasks
  for (int i = tid; i < BT * Hq; i += 256) {
    int m = i >> 8, n = i & 255;
    float hm = 0.5f * (bf2f(A0[m][8 + n]) + bf2f(A1[m][256 + n]));
    float cm = 0.5f * (c0[m][n] + c1[m][n]);
    for (int f = 0; f < Fq; ++f) {
      dec_h[(f * Bq + b0 + m) * Hq + n] = hm;
      dec_c[(f * Bq + b0 + m) * Hq + n] = cm;
    }
  }
  if (tid < BT * Fq) {
    int m = tid >> 3, k = tid & 7;
    dec_in[(b0 + m) * Fq + k] = x[((b0 + m) * TIN + (TIN - 1)) * Fq + k];
  }
}

// ---------------------------------------------------------------------------
// Decoder stage 1 (per step): LSTM cell + fc1 + qkv projection.
// grid = (64 batch tiles, 8 tasks). V is written TRANSPOSED (F,H,B) so the
// attention P@V GEMM gets dword-aligned B fragments.
// ---------------------------------------------------------------------------
__global__ __launch_bounds__(256) void dec_cell_kernel(
    const float* __restrict__ din,                   // (B,F)
    float* __restrict__ dec_h, float* __restrict__ dec_c,   // (F,B,H)
    const unsigned int* __restrict__ Pwhh,           // per-f stride 8*64*256
    const float* __restrict__ dWih,                  // (F,4H)
    const float* __restrict__ dbih, const float* __restrict__ dbhh, // (F,4H)
    const unsigned int* __restrict__ Pfc1,           // per-f stride 8*16*256
    const float* __restrict__ fc1b,                  // (F,H)
    const unsigned int* __restrict__ Pin,            // per-f stride 8*48*256
    const float* __restrict__ inb,                   // (F,3H)
    unsigned short* __restrict__ qb,                 // (F,B,H) bf16
    unsigned short* __restrict__ kb,                 // (F,B,H) bf16
    unsigned short* __restrict__ vt)                 // (F,H,B) bf16 (transposed)
{
  __shared__ unsigned short hA[BT][Hq];
  __shared__ float g[BT][G4];
  __shared__ unsigned short aA[BT][Hq];
  __shared__ float xi[BT];

  int tid = threadIdx.x, wave = tid >> 5, lane = tid & 31;
  int b0 = blockIdx.x * BT, f = blockIdx.y;

  for (int i = tid; i < BT * Hq; i += 256) {
    int m = i >> 8, n = i & 255;
    hA[m][n] = f2bf(dec_h[(f * Bq + b0 + m) * Hq + n]);
  }
  if (tid < BT) xi[tid] = din[(b0 + tid) * Fq + f];
  __syncthreads();

  // g = h @ Whh^T   (16 x 1024 x 256)
  v8f acc[8];
#pragma unroll
  for (int s = 0; s < 8; ++s) acc[s] = vzero8();
  gemm_panels<8>(&hA[0][0], Hq, 8, Pwhh + (long)f * (8 * 64 * 256), 64, wave * 8, acc);
  store_acc_lds<8>(&g[0][0], G4, wave * 8, acc, 1.f);
  __syncthreads();

  for (int i = tid; i < BT * Hq; i += 256) {
    int m = i >> 8, n = i & 255;
    float xv = xi[m];
    const float* bi = dbih + f * G4;
    const float* bh = dbhh + f * G4;
    const float* wi = dWih + f * G4;
    float gi = g[m][n]       + bi[n]       + bh[n]       + xv * wi[n];
    float gf = g[m][256 + n] + bi[256 + n] + bh[256 + n] + xv * wi[256 + n];
    float gg = g[m][512 + n] + bi[512 + n] + bh[512 + n] + xv * wi[512 + n];
    float go = g[m][768 + n] + bi[768 + n] + bh[768 + n] + xv * wi[768 + n];
    float c2 = sigf(gf) * dec_c[(f * Bq + b0 + m) * Hq + n] + sigf(gi) * tanhf(gg);
    float h2 = sigf(go) * tanhf(c2);
    dec_c[(f * Bq + b0 + m) * Hq + n] = c2;
    dec_h[(f * Bq + b0 + m) * Hq + n] = h2;
    hA[m][n] = f2bf(h2);
  }
  __syncthreads();

  // a = leaky(h2 @ fc1^T + b)   (16 x 256 x 256)
  v8f a2[2];
  a2[0] = vzero8(); a2[1] = vzero8();
  gemm_panels<2>(&hA[0][0], Hq, 8, Pfc1 + (long)f * (8 * 16 * 256), 16, wave * 2, a2);
  store_acc_lds<2>(&g[0][0], Hq, wave * 2, a2, 1.f);   // reuse g as 16x256
  __syncthreads();
  for (int i = tid; i < BT * Hq; i += 256) {
    int m = i >> 8, n = i & 255;
    aA[m][n] = f2bf(leaky((&g[0][0])[m * Hq + n] + fc1b[f * Hq + n]));
  }
  __syncthreads();

  // qkv = a @ in^T + b   (16 x 768 x 256) -> q/k row-major, v transposed
  v8f q6[6];
#pragma unroll
  for (int s = 0; s < 6; ++s) q6[s] = vzero8();
  gemm_panels<6>(&aA[0][0], Hq, 8, Pin + (long)f * (8 * 48 * 256), 48, wave * 6, q6);
  {
    int col = lane & 15, mh = (lane >> 4) * 8;
#pragma unroll
    for (int s = 0; s < 6; ++s) {
      int n = (wave * 6 + s) * 16 + col;
      int sec = n >> 8, nn = n & 255;
#pragma unroll
      for (int j = 0; j < 8; ++j) {
        float v = q6[s][j] + inb[f * 768 + n];
        unsigned short hv = f2bf(v);
        if (sec == 0)
          qb[((long)f * Bq + b0 + mh + j) * Hq + nn] = hv;
        else if (sec == 1)
          kb[((long)f * Bq + b0 + mh + j) * Hq + nn] = hv;
        else
          vt[((long)f * Hq + nn) * Bq + b0 + mh + j] = hv;
      }
    }
  }
}

// ---------------------------------------------------------------------------
// Decoder stage 2 (per step): cross-batch attention + attn_out + fc2.
// Each WG: 16 query rows vs all 1024 keys (flash-style, softmax in LDS).
// ---------------------------------------------------------------------------
__global__ __launch_bounds__(256) void dec_attn_kernel(
    const unsigned short* __restrict__ qbuf,          // (F,B,H) bf16
    const unsigned short* __restrict__ kbuf,          // (F,B,H) bf16
    const unsigned short* __restrict__ vtbuf,         // (F,H,B) bf16 transposed
    const unsigned int* __restrict__ Pout,            // per-f stride 8*16*256
    const float* __restrict__ outb,                   // (F,H)
    const float* __restrict__ f2W, const float* __restrict__ f2b, // (F,H),(F,1)
    float* __restrict__ din_next,                     // (B,F)
    float* __restrict__ out, int t)                   // (B,TOUT,F)
{
  __shared__ float S[BT][Bq];              // 64 KB scores
  __shared__ unsigned short Pp[BT][Bq];    // 32 KB softmax numerators (bf16)
  __shared__ unsigned short qA[BT][Hq];    // q tile, reused for O tile
  __shared__ float ao[BT][Hq];
  __shared__ float red[16][16];
  __shared__ float rowmax[16], rowinv[16];

  int tid = threadIdx.x, wave = tid >> 5, lane = tid & 31;
  int b0 = blockIdx.x * BT, f = blockIdx.y;
  const unsigned short* kp  = kbuf  + (long)f * Bq * Hq;
  const unsigned short* vtp = vtbuf + (long)f * Hq * Bq;

  for (int i = tid; i < BT * Hq; i += 256) {
    int m = i >> 8, n = i & 255;
    qA[m][n] = qbuf[(long)f * Bq * Hq + (long)(b0 + m) * Hq + n];
  }
  __syncthreads();

  // S = q @ k^T * (1/sqrt(H))     (16 x 1024 x 256), B-frags from global kbuf
  {
    v8f acc[8];
#pragma unroll
    for (int s = 0; s < 8; ++s) acc[s] = vzero8();
    constexpr int CH = chunk_of(8);
    gemm_core<8>(&qA[0][0], Hq, 8,
        [&](Frag* bf, int kt, int s0) {
          int ln = threadIdx.x & 31;
          int row = ln & 15;
#pragma unroll
          for (int s = 0; s < CH; ++s) {
            int keyb = (wave * 8 + s0 + s) * 16;
#pragma unroll
            for (int j = 0; j < 8; ++j)
              bf[s].u[j] = *(const unsigned int*)(kp + (long)(keyb + row) * Hq +
                                                  kt * 32 + frag_kB(ln, j, 0));
          }
        },
        acc);
    store_acc_lds<8>(&S[0][0], Bq, wave * 8, acc, 0.0625f);
  }
  __syncthreads();

  // row softmax: 16 threads per row
  {
    int row = tid & 15, chunk = tid >> 4;
    float mx = -1e30f;
    for (int k2 = chunk * 64; k2 < chunk * 64 + 64; ++k2) mx = fmaxf(mx, S[row][k2]);
    red[row][chunk] = mx;
    __syncthreads();
    if (chunk == 0) {
      float m2 = red[row][0];
      for (int c = 1; c < 16; ++c) m2 = fmaxf(m2, red[row][c]);
      rowmax[row] = m2;
    }
    __syncthreads();
    float rm = rowmax[row], sm = 0.f;
    for (int k2 = chunk * 64; k2 < chunk * 64 + 64; ++k2) {
      float e = __expf(S[row][k2] - rm);
      sm += e;
      Pp[row][k2] = f2bf(e);
    }
    red[row][chunk] = sm;
    __syncthreads();
    if (chunk == 0) {
      float s2 = 0.f;
      for (int c = 0; c < 16; ++c) s2 += red[row][c];
      rowinv[row] = 1.f / s2;
    }
    __syncthreads();
  }

  // O = P @ V (16 x 256 x 1024), 1/rowsum folded into epilogue; O -> qA (bf16)
  {
    v8f o2[2];
    o2[0] = vzero8(); o2[1] = vzero8();
    constexpr int CH = chunk_of(2);
    gemm_core<2>(&Pp[0][0], Bq, 32,
        [&](Frag* bf, int kt, int s0) {
          int ln = threadIdx.x & 31;
          int col = ln & 15;
#pragma unroll
          for (int s = 0; s < CH; ++s) {
            int n = (wave * 2 + s0 + s) * 16 + col;
#pragma unroll
            for (int j = 0; j < 8; ++j)
              bf[s].u[j] = *(const unsigned int*)(vtp + (long)n * Bq +
                                                  kt * 32 + frag_kB(ln, j, 0));
          }
        },
        o2);
    int col = lane & 15, mh = (lane >> 4) * 8;
#pragma unroll
    for (int s = 0; s < 2; ++s) {
      int n = (wave * 2 + s) * 16 + col;
#pragma unroll
      for (int j = 0; j < 8; ++j)
        qA[mh + j][n] = f2bf(o2[s][j] * rowinv[mh + j]);
    }
  }
  __syncthreads();

  // ao = O @ attn_out^T + b   (16 x 256 x 256)
  {
    v8f c2a[2];
    c2a[0] = vzero8(); c2a[1] = vzero8();
    gemm_panels<2>(&qA[0][0], Hq, 8, Pout + (long)f * (8 * 16 * 256), 16, wave * 2, c2a);
    int col = lane & 15, mh = (lane >> 4) * 8;
#pragma unroll
    for (int s = 0; s < 2; ++s) {
      int n = (wave * 2 + s) * 16 + col;
#pragma unroll
      for (int j = 0; j < 8; ++j)
        ao[mh + j][n] = c2a[s][j] + outb[f * Hq + n];
    }
  }
  __syncthreads();

  // out = leaky(ao @ f2W^T + f2b)  -> next decoder input + final output
  if (tid < BT) {
    int m = tid;
    float s = f2b[f];
    for (int k2 = 0; k2 < Hq; ++k2) s += ao[m][k2] * f2W[f * Hq + k2];
    float v = leaky(s);
    din_next[(b0 + m) * Fq + f] = v;
    out[((b0 + m) * TOUT + t) * Fq + f] = v;
  }
}

// ---------------------------------------------------------------------------
// Host launcher
// ---------------------------------------------------------------------------
extern "C" void kernel_launch(void* const* d_in, const int* in_sizes, int n_in,
                              void* d_out, int out_size, void* d_ws, size_t ws_size,
                              hipStream_t stream) {
  const float* x        = (const float*)d_in[0];
  const float* eWih0    = (const float*)d_in[1];
  const float* eWhh0    = (const float*)d_in[2];
  const float* ebih0    = (const float*)d_in[3];
  const float* ebhh0    = (const float*)d_in[4];
  const float* eWih1    = (const float*)d_in[5];
  const float* eWhh1    = (const float*)d_in[6];
  const float* ebih1    = (const float*)d_in[7];
  const float* ebhh1    = (const float*)d_in[8];
  const float* dWih     = (const float*)d_in[9];
  const float* dWhh     = (const float*)d_in[10];
  const float* dbih     = (const float*)d_in[11];
  const float* dbhh     = (const float*)d_in[12];
  const float* fc1W     = (const float*)d_in[13];
  const float* fc1b     = (const float*)d_in[14];
  const float* inW      = (const float*)d_in[15];
  const float* inb      = (const float*)d_in[16];
  const float* outW     = (const float*)d_in[17];
  const float* outb     = (const float*)d_in[18];
  const float* f2W      = (const float*)d_in[19];
  const float* f2b      = (const float*)d_in[20];
  float* outp = (float*)d_out;

  // workspace carve-up (256B aligned)
  char* ws = (char*)d_ws;
  size_t off = 0;
  auto take = [&](size_t bytes) -> char* {
    char* p = ws + off;
    off = (off + bytes + 255) & ~(size_t)255;
    return p;
  };
  unsigned int* P0   = (unsigned int*)take((size_t)9  * 64 * 256 * 4);
  unsigned int* P1   = (unsigned int*)take((size_t)16 * 64 * 256 * 4);
  unsigned int* Pwhh = (unsigned int*)take((size_t)Fq * 8 * 64 * 256 * 4);
  unsigned int* Pfc1 = (unsigned int*)take((size_t)Fq * 8 * 16 * 256 * 4);
  unsigned int* Pin  = (unsigned int*)take((size_t)Fq * 8 * 48 * 256 * 4);
  unsigned int* Pout = (unsigned int*)take((size_t)Fq * 8 * 16 * 256 * 4);
  float* dec_h = (float*)take((size_t)Fq * Bq * Hq * 4);
  float* dec_c = (float*)take((size_t)Fq * Bq * Hq * 4);
  float* din0  = (float*)take((size_t)Bq * Fq * 4);
  float* din1  = (float*)take((size_t)Bq * Fq * 4);
  unsigned short* qb = (unsigned short*)take((size_t)Fq * Bq * Hq * 2);
  unsigned short* kb = (unsigned short*)take((size_t)Fq * Bq * Hq * 2);
  unsigned short* vt = (unsigned short*)take((size_t)Fq * Hq * Bq * 2);
  if (off > ws_size) return;  // workspace too small: deterministic no-op

  // ---- weight packing (bf16 B-fragment panels) ----
  pack_panels_kernel<<<dim3((9 * 64 * 256 + 255) / 256, 1), 256, 0, stream>>>(
      eWih0, 8, eWhh0, 256, 0, P0, 0, 64, 9);
  pack_panels_kernel<<<dim3((16 * 64 * 256 + 255) / 256, 1), 256, 0, stream>>>(
      eWih1, 256, eWhh1, 256, 0, P1, 0, 64, 16);
  pack_panels_kernel<<<dim3((8 * 64 * 256 + 255) / 256, Fq), 256, 0, stream>>>(
      nullptr, 0, dWhh, 256, (long)G4 * Hq, Pwhh, (long)8 * 64 * 256, 64, 8);
  pack_panels_kernel<<<dim3((8 * 16 * 256 + 255) / 256, Fq), 256, 0, stream>>>(
      nullptr, 0, fc1W, 256, (long)Hq * Hq, Pfc1, (long)8 * 16 * 256, 16, 8);
  pack_panels_kernel<<<dim3((8 * 48 * 256 + 255) / 256, Fq), 256, 0, stream>>>(
      nullptr, 0, inW, 256, (long)3 * Hq * Hq, Pin, (long)8 * 48 * 256, 48, 8);
  pack_panels_kernel<<<dim3((8 * 16 * 256 + 255) / 256, Fq), 256, 0, stream>>>(
      nullptr, 0, outW, 256, (long)Hq * Hq, Pout, (long)8 * 16 * 256, 16, 8);

  // ---- encoder (persistent, 168 steps fused in-kernel) ----
  encoder_kernel<<<NBT, 256, 0, stream>>>(x, P0, P1, ebih0, ebhh0, ebih1, ebhh1,
                                          dec_h, dec_c, din0);

  // ---- decoder: 24 sequential steps, 2 launches each ----
  float* cur = din0;
  float* nxt = din1;
  for (int t = 0; t < TOUT; ++t) {
    dec_cell_kernel<<<dim3(NBT, Fq), 256, 0, stream>>>(
        cur, dec_h, dec_c, Pwhh, dWih, dbih, dbhh, Pfc1, fc1b, Pin, inb,
        qb, kb, vt);
    dec_attn_kernel<<<dim3(NBT, Fq), 256, 0, stream>>>(
        qb, kb, vt, Pout, outb, f2W, f2b, nxt, outp, t);
    float* tmp = cur; cur = nxt; nxt = tmp;
  }
  (void)in_sizes; (void)n_in; (void)out_size;
}
